// MultiHeadOppAttention_65953517797858
// MI455X (gfx1250) — matrix operand order repaired and checked by reading, not compiled, not run on hardware
//
#include <hip/hip_runtime.h>
#include <cstdint>

// ---------------------------------------------------------------------------
// MultiHeadOppAttention for MI455X (gfx1250), bf16 WMMA pipeline.
//   S=512, A=O=128, D=256, H=8, dk=dv=64, C=H*dk=512, E=512
// Pipeline:
//   k0: convert/transpose weights fp32 -> bf16  (Wt[j][d], Wo_t[e][r])
//   k1: projection GEMM  X(65536x256) x Wt^T -> scatter-store scrambled
//       K:(s,a,512)  Q:(s,o,512)  V transposed:(s,c,128)
//   k2: per-s fused attention: compat WMMA + softmax + attn@V WMMA -> heads bf16
//   k3: output GEMM heads(65536x512) x W_out(512x512) -> fp32 out
// ---------------------------------------------------------------------------

typedef __bf16 bf16;
typedef bf16  v16bf __attribute__((ext_vector_type(16)));
typedef float v8f   __attribute__((ext_vector_type(8)));

#define WMMA_BF16(a, b, c) \
  __builtin_amdgcn_wmma_f32_16x16x32_bf16(false, (a), false, (b), (short)0, (c), false, false)

union frag_u { uint4 q[2]; v16bf v; };

// A-matrix 16x32 bf16 layout (ISA 7.12.2): lane = halfK*16 + m,
// elements 0..7 -> K = half*8 + e, elements 8..15 -> K = 16 + half*8 + (e-8).
// Row-major source => two contiguous 16B loads per lane.
__device__ __forceinline__ v16bf load_a_bf(const bf16* __restrict__ row, int kc) {
  const int half = (threadIdx.x >> 4) & 1;
  frag_u u;
  u.q[0] = *reinterpret_cast<const uint4*>(row + kc + half * 8);
  u.q[1] = *reinterpret_cast<const uint4*>(row + kc + 16 + half * 8);
  return u.v;
}

// B-matrix 32x16 bf16 layout: lane = halfK*16 + n, same K order as A:
// elements 0..7 -> K = half*8 + e, elements 8..15 -> K = 16 + half*8 + (e-8).
__device__ __forceinline__ v16bf load_b_bf(const bf16* __restrict__ row, int kc) {
  const int half = (threadIdx.x >> 4) & 1;
  frag_u u;
  u.q[0] = *reinterpret_cast<const uint4*>(row + kc + half * 8);
  u.q[1] = *reinterpret_cast<const uint4*>(row + kc + 16 + half * 8);
  return u.v;
}

// A-fragment from fp32 row (on-the-fly fp32 -> bf16 conversion).
__device__ __forceinline__ v16bf load_a_f32(const float* __restrict__ row, int kc) {
  const int half = (threadIdx.x >> 4) & 1;
  const float* p0 = row + kc + half * 8;
  const float* p1 = row + kc + 16 + half * 8;
  v16bf f;
#pragma unroll
  for (int i = 0; i < 8; ++i) {
    f[i]     = (bf16)p0[i];
    f[8 + i] = (bf16)p1[i];
  }
  return f;
}

__device__ __forceinline__ v8f zero_v8f() {
  v8f z = {0.f, 0.f, 0.f, 0.f, 0.f, 0.f, 0.f, 0.f};
  return z;
}

// ---------------------------------------------------------------------------
// k0: weight convert + transpose.
//   Wt_x[j*256 + d] = Wx[h][d][k],  j = h*64+k   (so B-frags read contiguous d)
//   Wo_t[e*512 + r] = W_out_flat[r*512 + e]      (so B-frags read contiguous r)
// ---------------------------------------------------------------------------
__global__ void convert_weights(const float* __restrict__ Wq, const float* __restrict__ Wk,
                                const float* __restrict__ Wv, const float* __restrict__ Wo,
                                bf16* __restrict__ Wtq, bf16* __restrict__ Wtk,
                                bf16* __restrict__ Wtv, bf16* __restrict__ Wot) {
  int i = blockIdx.x * blockDim.x + threadIdx.x;
  if (i < 512 * 256) {
    int j = i >> 8, d = i & 255;
    int h = j >> 6, k = j & 63;
    int src = h * (256 * 64) + d * 64 + k;
    Wtq[i] = (bf16)Wq[src];
    Wtk[i] = (bf16)Wk[src];
    Wtv[i] = (bf16)Wv[src];
  }
  if (i < 512 * 512) {
    int e = i >> 9, r = i & 511;
    Wot[i] = (bf16)Wo[r * 512 + e];
  }
}

// ---------------------------------------------------------------------------
// k1: projection GEMM, M=65536, K=256, N=512.  Block = 16 rows; wave w owns
// 64 columns (4 accum tiles). Scatter-store resolves the head-major reshape:
//   proj[h, n, k] lands at (s = h*64 + n/1024, row = (n%1024)/8, c = (n%8)*64+k)
// vmode 0: Out[(s*128 + row)*512 + c]   (K and Q, row-major (s,row,c))
// vmode 1: Out[(s*512 + c)*128 + row]   (V transposed: (s,c,o))
// ---------------------------------------------------------------------------
__global__ __launch_bounds__(256) void proj_kernel(const float* __restrict__ X,
                                                   const bf16* __restrict__ Wt,
                                                   bf16* __restrict__ Out, int vmode) {
  const int lane = threadIdx.x & 31;
  const int wave = threadIdx.x >> 5;
  const int r    = lane & 15;
  const int half = lane >> 4;
  const int m0 = blockIdx.x * 16;
  const int n0 = wave * 64;

  v8f acc[4];
#pragma unroll
  for (int t = 0; t < 4; ++t) acc[t] = zero_v8f();

  const float* arow = X + (size_t)(m0 + r) * 256;
#pragma unroll
  for (int kc = 0; kc < 256; kc += 32) {
    v16bf a = load_a_f32(arow, kc);
#pragma unroll
    for (int t = 0; t < 4; ++t) {
      v16bf b = load_b_bf(Wt + (size_t)(n0 + t * 16 + r) * 256, kc);
      acc[t] = WMMA_BF16(a, b, acc[t]);
    }
  }

#pragma unroll
  for (int t = 0; t < 4; ++t) {
    const int j = n0 + t * 16 + r;    // C/D layout: N = lane%16
    const int h = j >> 6, k = j & 63;
#pragma unroll
    for (int v = 0; v < 8; ++v) {
      const int m   = m0 + v + 8 * half;  // C/D layout: M = vgpr + 8*halfwave
      const int s   = h * 64 + (m >> 10);
      const int rem = m & 1023;
      const int aa  = rem >> 3;
      const int chi = rem & 7;
      const int c   = chi * 64 + k;
      const size_t idx = vmode ? ((size_t)s * 512 + c) * 128 + aa
                               : ((size_t)s * 128 + aa) * 512 + c;
      Out[idx] = (bf16)acc[t][v];
    }
  }
}

// ---------------------------------------------------------------------------
// k2: fused attention, one block per s. Wave w owns rows a0 = w*16.
//   compat = K_s (128x512) x Q_s^T (norm folded in), row softmax in registers
//   (shfl_xor within 16-lane half-groups matches the C-matrix layout), attn
//   staged bf16 through LDS to become A-fragments, then attn x V_s^T.
// ---------------------------------------------------------------------------
__global__ __launch_bounds__(256) void attn_kernel(const bf16* __restrict__ Kb,
                                                   const bf16* __restrict__ Qb,
                                                   const bf16* __restrict__ Vt,
                                                   bf16* __restrict__ Hd) {
  __shared__ __align__(16) bf16 lds_attn[8][16][128];  // 32 KB
  const int s    = blockIdx.x;
  const int lane = threadIdx.x & 31;
  const int wave = threadIdx.x >> 5;
  const int r    = lane & 15;
  const int half = lane >> 4;
  const int a0   = wave * 16;

  const bf16* Ks = Kb + (size_t)s * 128 * 512;
  const bf16* Qs = Qb + (size_t)s * 128 * 512;
  const bf16* Vs = Vt + (size_t)s * 512 * 128;

  // --- compat: 16 x 128 fp32 per wave, K over C=512 ---
  v8f acc[8];
#pragma unroll
  for (int t = 0; t < 8; ++t) acc[t] = zero_v8f();

  const bf16* arow = Ks + (size_t)(a0 + r) * 512;
#pragma unroll 4
  for (int kc = 0; kc < 512; kc += 32) {
    v16bf a = load_a_bf(arow, kc);
#pragma unroll
    for (int t = 0; t < 8; ++t) {
      v16bf b = load_b_bf(Qs + (size_t)(t * 16 + r) * 512, kc);
      acc[t] = WMMA_BF16(a, b, acc[t]);
    }
  }

  // --- row softmax (norm = 1/sqrt(64)); row = v + 8*half, cols across lanes ---
  const float norm = 0.125f;
  float rinv[8];
#pragma unroll
  for (int v = 0; v < 8; ++v) {
    float mx = -3.4e38f;
#pragma unroll
    for (int t = 0; t < 8; ++t) mx = fmaxf(mx, acc[t][v]);
    mx = fmaxf(mx, __shfl_xor(mx, 1, 32));
    mx = fmaxf(mx, __shfl_xor(mx, 2, 32));
    mx = fmaxf(mx, __shfl_xor(mx, 4, 32));
    mx = fmaxf(mx, __shfl_xor(mx, 8, 32));
    mx *= norm;
    float sum = 0.f;
#pragma unroll
    for (int t = 0; t < 8; ++t) {
      float p = __expf(acc[t][v] * norm - mx);
      acc[t][v] = p;
      sum += p;
    }
    sum += __shfl_xor(sum, 1, 32);
    sum += __shfl_xor(sum, 2, 32);
    sum += __shfl_xor(sum, 4, 32);
    sum += __shfl_xor(sum, 8, 32);
    rinv[v] = 1.0f / sum;
  }

  // stage attn (bf16) to LDS in row-major so it reloads as A-fragments
#pragma unroll
  for (int t = 0; t < 8; ++t)
#pragma unroll
    for (int v = 0; v < 8; ++v)
      lds_attn[wave][v + 8 * half][t * 16 + r] = (bf16)(acc[t][v] * rinv[v]);
  __syncthreads();

  // --- heads = attn (16x128) x V_s (128x512), N in 4 chunks of 128 ---
  const bf16* prow = &lds_attn[wave][r][0];
#pragma unroll
  for (int nc = 0; nc < 4; ++nc) {
    v8f h[8];
#pragma unroll
    for (int t = 0; t < 8; ++t) h[t] = zero_v8f();
#pragma unroll
    for (int kc = 0; kc < 128; kc += 32) {
      v16bf a = load_a_bf(prow, kc);
#pragma unroll
      for (int t = 0; t < 8; ++t) {
        v16bf b = load_b_bf(Vs + (size_t)(nc * 128 + t * 16 + r) * 128, kc);
        h[t] = WMMA_BF16(a, b, h[t]);
      }
    }
#pragma unroll
    for (int t = 0; t < 8; ++t)
#pragma unroll
      for (int v = 0; v < 8; ++v)
        Hd[((size_t)s * 128 + (a0 + v + 8 * half)) * 512 + nc * 128 + t * 16 + r] =
            (bf16)h[t][v];
  }
}

// ---------------------------------------------------------------------------
// k3: output GEMM, M=65536, K=512, N=512, fp32 result.
// ---------------------------------------------------------------------------
__global__ __launch_bounds__(256) void out_kernel(const bf16* __restrict__ Hd,
                                                  const bf16* __restrict__ Wot,
                                                  float* __restrict__ Out) {
  const int lane = threadIdx.x & 31;
  const int wave = threadIdx.x >> 5;
  const int r    = lane & 15;
  const int half = lane >> 4;
  const int m0 = blockIdx.x * 16;
  const int n0 = wave * 64;

  v8f acc[4];
#pragma unroll
  for (int t = 0; t < 4; ++t) acc[t] = zero_v8f();

  const bf16* arow = Hd + (size_t)(m0 + r) * 512;
#pragma unroll 4
  for (int kc = 0; kc < 512; kc += 32) {
    v16bf a = load_a_bf(arow, kc);
#pragma unroll
    for (int t = 0; t < 4; ++t) {
      v16bf b = load_b_bf(Wot + (size_t)(n0 + t * 16 + r) * 512, kc);
      acc[t] = WMMA_BF16(a, b, acc[t]);
    }
  }

#pragma unroll
  for (int t = 0; t < 4; ++t)
#pragma unroll
    for (int v = 0; v < 8; ++v)
      Out[(size_t)(m0 + v + 8 * half) * 512 + n0 + t * 16 + r] = acc[t][v];
}

// ---------------------------------------------------------------------------
extern "C" void kernel_launch(void* const* d_in, const int* in_sizes, int n_in,
                              void* d_out, int out_size, void* d_ws, size_t ws_size,
                              hipStream_t stream) {
  const float* h    = (const float*)d_in[0];
  const float* hOpp = (const float*)d_in[1];
  const float* Wq   = (const float*)d_in[2];
  const float* Wk   = (const float*)d_in[3];
  const float* Wv   = (const float*)d_in[4];
  const float* Wo   = (const float*)d_in[5];
  float* out = (float*)d_out;

  // Workspace layout (bf16 intermediates; ~257 MiB total)
  char* ws = (char*)d_ws;
  constexpr size_t SZ_BIG = (size_t)512 * 128 * 512 * sizeof(bf16);  // 64 MiB
  constexpr size_t SZ_W   = (size_t)512 * 256 * sizeof(bf16);        // 256 KiB
  bf16* Qb  = (bf16*)(ws);
  bf16* Kb  = (bf16*)(ws + SZ_BIG);
  bf16* Vt  = (bf16*)(ws + 2 * SZ_BIG);
  bf16* Hd  = (bf16*)(ws + 3 * SZ_BIG);
  bf16* Wtq = (bf16*)(ws + 4 * SZ_BIG);
  bf16* Wtk = (bf16*)(ws + 4 * SZ_BIG + SZ_W);
  bf16* Wtv = (bf16*)(ws + 4 * SZ_BIG + 2 * SZ_W);
  bf16* Wot = (bf16*)(ws + 4 * SZ_BIG + 3 * SZ_W);

  convert_weights<<<1024, 256, 0, stream>>>(Wq, Wk, Wv, Wo, Wtq, Wtk, Wtv, Wot);
  proj_kernel<<<4096, 256, 0, stream>>>(h,    Wtk, Kb, 0);
  proj_kernel<<<4096, 256, 0, stream>>>(hOpp, Wtq, Qb, 0);
  proj_kernel<<<4096, 256, 0, stream>>>(hOpp, Wtv, Vt, 1);
  attn_kernel<<<512, 256, 0, stream>>>(Kb, Qb, Vt, Hd);
  out_kernel<<<4096, 256, 0, stream>>>(Hd, Wot, out);
}
